// WaveNetModel_11398843203666
// MI455X (gfx1250) — matrix operand order repaired and checked by reading, not compiled, not run on hardware
//
#include <hip/hip_runtime.h>
#include <hip/hip_fp16.h>

typedef __attribute__((ext_vector_type(16))) _Float16 v16h;
typedef __attribute__((ext_vector_type(8)))  _Float16 v8h;
typedef __attribute__((ext_vector_type(8)))  float    v8f;
typedef __attribute__((ext_vector_type(4)))  float    v4f;

#define NSTAGES 40
#define NB      8
#define LIN     8192
#define POSCAP  8224      // 8192 + zero tail for tap/pad reads (= 257*32)
#define RC      32
#define DC      32
#define SC      256
#define EC      256
#define NCLASS  256
#define OUTLEN  4096
#define WINBASE 4095      // final skip length is 8191; we keep [4095, 8191)

// A-fragment blob tile offsets (1 tile = 16x32 f16 A fragment = 1 KB)
#define OFF_START 0
#define OFF_FILT  16
#define OFF_GATE  176
#define OFF_RES   336
#define OFF_SKIP  416
#define OFF_END1  1056
#define OFF_END2  1184
#define NTILES    1312

// ---------------- device helpers ----------------

__device__ __forceinline__ v8f wmma_f16(v16h a, v16h b, v8f c) {
  // D = A(16x32 f16) x B(32x16 f16) + C(16x16 f32)
  return __builtin_amdgcn_wmma_f32_16x16x32_f16(false, a, false, b, (short)0, c,
                                                false, false);
}

// K index for the 16-bit A-matrix fragment (contiguous, matches B fragments):
// lanes 0-15 hold K 0..15, lanes 16-31 hold K 16..31
__device__ __forceinline__ int a_k(int lane, int h) {
  return ((lane >> 4) << 4) + h;
}

__device__ __forceinline__ v16h load_afrag(const _Float16* blob, int tile) {
  int lane = threadIdx.x & 31;
  const v8h* q = (const v8h*)(blob + (size_t)tile * 512 + lane * 16);
  v8h lo = q[0], hi = q[1];
  v16h r;
#pragma unroll
  for (int i = 0; i < 8; ++i) { r[i] = lo[i]; r[8 + i] = hi[i]; }
  return r;
}

// B fragment (32 K-values x 16 cols), straight layout:
// lanes 0-15 = K 0..15 (col = lane), lanes 16-31 = K 16..31
__device__ __forceinline__ v16h load_bfrag32(const _Float16* p) {
  int base16 = ((threadIdx.x >> 4) & 1) * 16;
  const v8h* q = (const v8h*)(p + base16);
  v8h lo = q[0], hi = q[1];
  v16h r;
#pragma unroll
  for (int i = 0; i < 8; ++i) { r[i] = lo[i]; r[8 + i] = hi[i]; }
  return r;
}

__device__ __forceinline__ v16h bfrag_from_f32_relu(const float* p) {
  int base16 = ((threadIdx.x >> 4) & 1) * 16;
  const v4f* q = (const v4f*)(p + base16);
  v16h r;
#pragma unroll
  for (int j = 0; j < 4; ++j) {
    v4f v = q[j];
#pragma unroll
    for (int i = 0; i < 4; ++i) r[j * 4 + i] = (_Float16)fmaxf(v[i], 0.0f);
  }
  return r;
}

// Two 16-row D tiles (channels 0..15 / 16..31) -> one 32-K B fragment.
// Only a lane xor-16 exchange is needed; layouts otherwise compose.
__device__ __forceinline__ v16h dpair_to_bfrag(v8f d0, v8f d1) {
  bool lo = (threadIdx.x & 31) < 16;
  v16h r;
#pragma unroll
  for (int h = 0; h < 8; ++h) {
    float s0 = __shfl_xor(d0[h], 16, 32);
    float s1 = __shfl_xor(d1[h], 16, 32);
    r[h]     = (_Float16)(lo ? d0[h] : s1);
    r[8 + h] = (_Float16)(lo ? s0 : d1[h]);
  }
  return r;
}

__device__ __forceinline__ void store8_f32(float* p, v8f d) {
  v4f a, b;
#pragma unroll
  for (int i = 0; i < 4; ++i) { a[i] = d[i]; b[i] = d[4 + i]; }
  ((v4f*)p)[0] = a; ((v4f*)p)[1] = b;
}

__device__ __forceinline__ void store8_f16(_Float16* p, v8f d) {
  v8h h;
#pragma unroll
  for (int i = 0; i < 8; ++i) h[i] = (_Float16)d[i];
  *(v8h*)p = h;
}

__device__ __forceinline__ v8f load8_f32(const float* p) {
  v4f a = ((const v4f*)p)[0], b = ((const v4f*)p)[1];
  v8f r;
#pragma unroll
  for (int i = 0; i < 4; ++i) { r[i] = a[i]; r[4 + i] = b[i]; }
  return r;
}

// ---------------- prep kernels ----------------

// Pack all weights (f32) into per-lane A-fragment f16 blob.
__global__ void prep_weights(const float* start_w, const float* filter_w,
                             const float* gate_w, const float* res_w,
                             const float* skip_w, const float* end1_w,
                             const float* end2_w, _Float16* blob) {
  int t = blockIdx.x;          // tile
  int lane = threadIdx.x;      // blockDim = 32
  int r = lane & 15;
  _Float16* dst = blob + (size_t)t * 512 + lane * 16;
#pragma unroll
  for (int h = 0; h < 16; ++h) {
    int kk = a_k(lane, h);
    float v;
    if (t < OFF_FILT) {
      int u = t; int kg = u >> 1, mt = u & 1;
      v = start_w[(mt * 16 + r) * NCLASS + kg * 32 + kk];
    } else if (t < OFF_GATE) {
      int u = t - OFF_FILT; int s = u >> 2, tap = (u >> 1) & 1, mt = u & 1;
      v = filter_w[(((s * DC) + (mt * 16 + r)) * RC + kk) * 2 + tap];
    } else if (t < OFF_RES) {
      int u = t - OFF_GATE; int s = u >> 2, tap = (u >> 1) & 1, mt = u & 1;
      v = gate_w[(((s * DC) + (mt * 16 + r)) * RC + kk) * 2 + tap];
    } else if (t < OFF_SKIP) {
      int u = t - OFF_RES; int s = u >> 1, mt = u & 1;
      v = res_w[(s * RC + mt * 16 + r) * DC + kk];
    } else if (t < OFF_END1) {
      int u = t - OFF_SKIP; int s = u >> 4, mt = u & 15;
      v = skip_w[(s * SC + mt * 16 + r) * DC + kk];
    } else if (t < OFF_END2) {
      int u = t - OFF_END1; int mt = u >> 3, kg = u & 7;
      v = end1_w[(mt * 16 + r) * SC + kg * 32 + kk];
    } else {
      int u = t - OFF_END2; int mt = u >> 3, kg = u & 7;
      v = end2_w[(mt * 16 + r) * EC + kg * 32 + kk];
    }
    dst[h] = (_Float16)v;
  }
}

// (N, C, L) f32 -> (N, L, C) f16 for WMMA B-operand loads.
__global__ void prep_input(const float* __restrict__ in,
                           _Float16* __restrict__ out, int total) {
  int i = blockIdx.x * blockDim.x + threadIdx.x;
  if (i >= total) return;
  int pos = i % LIN;
  int rem = i / LIN;
  int c = rem % NCLASS;
  int n = rem / NCLASS;
  out[((size_t)(n * LIN + pos)) * NCLASS + c] = (_Float16)in[i];
}

// ---------------- start conv: x = start_w @ input ----------------

__global__ void __launch_bounds__(256)
start_kernel(const _Float16* __restrict__ in16, const _Float16* __restrict__ blob,
             float* __restrict__ x32, _Float16* __restrict__ x16) {
  int lane = threadIdx.x & 31;
  int flat = blockIdx.x * (blockDim.x >> 5) + (threadIdx.x >> 5);
  int tile = flat % 514;
  int n = flat / 514;
  if (n >= NB) return;
  int col = lane & 15, half8 = ((lane >> 4) & 1) * 8;
  int tb = tile * 16, p = tb + col;
  float*    o32 = x32 + ((size_t)(n * POSCAP + p)) * 32;
  _Float16* o16 = x16 + ((size_t)(n * POSCAP + p)) * 32;
  if (tb >= LIN) {  // zero tail [8192, 8224)
    v8f z = {};
#pragma unroll
    for (int mt = 0; mt < 2; ++mt) {
      store8_f32(o32 + mt * 16 + half8, z);
      store8_f16(o16 + mt * 16 + half8, z);
    }
    return;
  }
  const _Float16* row = in16 + ((size_t)(n * LIN + p)) * NCLASS;
  v16h b[8];
#pragma unroll
  for (int kg = 0; kg < 8; ++kg) b[kg] = load_bfrag32(row + kg * 32);
#pragma unroll
  for (int mt = 0; mt < 2; ++mt) {
    v8f d = {};
#pragma unroll
    for (int kg = 0; kg < 8; ++kg)
      d = wmma_f16(load_afrag(blob, OFF_START + kg * 2 + mt), b[kg], d);
    store8_f32(o32 + mt * 16 + half8, d);
    store8_f16(o16 + mt * 16 + half8, d);
  }
}

// ---------------- fused WaveNet stage (2 position-tiles per wave) ----------
// out[p] = res_w @ (tanh(Wf*res)*sig(Wg*res)) + res[p+1];  skip += skip_w @ x
// Each wave covers 32 positions so every A fragment feeds two WMMAs.

__global__ void __launch_bounds__(256)
stage_kernel(const float* __restrict__ xin32, const _Float16* __restrict__ xin16,
             float* __restrict__ xout32, _Float16* __restrict__ xout16,
             float* __restrict__ skipwin, const _Float16* __restrict__ blob,
             int stage, int Lout) {
  int lane = threadIdx.x & 31;
  int flat = blockIdx.x * (blockDim.x >> 5) + (threadIdx.x >> 5);
  int pair = flat % 257;            // 257 * 32 = 8224 = POSCAP
  int n = flat / 257;
  if (n >= NB) return;
  int col = lane & 15, half8 = ((lane >> 4) & 1) * 8;
  int tb = pair * 32;
  int p[2] = {tb + col, tb + 16 + col};

  float*    o32[2];
  _Float16* o16[2];
#pragma unroll
  for (int j = 0; j < 2; ++j) {
    o32[j] = xout32 + ((size_t)(n * POSCAP + p[j])) * 32;
    o16[j] = xout16 + ((size_t)(n * POSCAP + p[j])) * 32;
  }

  if (tb >= Lout) {  // fully invalid pair: maintain zero invariant
    v8f z = {};
#pragma unroll
    for (int j = 0; j < 2; ++j)
#pragma unroll
      for (int mt = 0; mt < 2; ++mt) {
        store8_f32(o32[j] + mt * 16 + half8, z);
        store8_f16(o16[j] + mt * 16 + half8, z);
      }
    return;
  }

  // B fragments for both sub-tiles and both conv taps (pad region reads 0)
  v16h b0[2], b1[2];
#pragma unroll
  for (int j = 0; j < 2; ++j) {
    const _Float16* rowin = xin16 + ((size_t)(n * POSCAP + p[j])) * 32;
    b0[j] = load_bfrag32(rowin);
    b1[j] = load_bfrag32(rowin + 32);
  }

  // filter/gate GEMMs + gated activation; A fragments reused across sub-tiles
  v8f xg[2][2];
#pragma unroll
  for (int mt = 0; mt < 2; ++mt) {
    v16h af0 = load_afrag(blob, OFF_FILT + stage * 4 + mt);
    v16h af1 = load_afrag(blob, OFF_FILT + stage * 4 + 2 + mt);
    v16h ag0 = load_afrag(blob, OFF_GATE + stage * 4 + mt);
    v16h ag1 = load_afrag(blob, OFF_GATE + stage * 4 + 2 + mt);
#pragma unroll
    for (int j = 0; j < 2; ++j) {
      v8f z = {};
      v8f f = wmma_f16(af0, b0[j], z);
      f     = wmma_f16(af1, b1[j], f);
      v8f g = wmma_f16(ag0, b0[j], z);
      g     = wmma_f16(ag1, b1[j], g);
      v8f xv;
#pragma unroll
      for (int i = 0; i < 8; ++i)
        xv[i] = tanhf(f[i]) * (1.0f / (1.0f + __expf(-g[i])));
      xg[j][mt] = xv;
    }
  }
  v16h xb[2] = {dpair_to_bfrag(xg[0][0], xg[0][1]),
                dpair_to_bfrag(xg[1][0], xg[1][1])};

  // skip accumulation over the output window [WINBASE, Lout)
  bool inwin[2] = {(p[0] >= WINBASE) && (p[0] < Lout),
                   (p[1] >= WINBASE) && (p[1] < Lout)};
  if (tb + 31 >= WINBASE) {  // wave-uniform guard
    float* srow0 = skipwin + ((size_t)n * OUTLEN + (p[0] - WINBASE)) * SC;
    float* srow1 = srow0 + 16 * SC;
#pragma unroll
    for (int mt = 0; mt < 16; ++mt) {
      v16h a = load_afrag(blob, OFF_SKIP + stage * 16 + mt);
      v8f c0 = {};
      if (inwin[0]) c0 = load8_f32(srow0 + mt * 16 + half8);
      v8f d0 = wmma_f16(a, xb[0], c0);
      if (inwin[0]) store8_f32(srow0 + mt * 16 + half8, d0);
      v8f c1 = {};
      if (inwin[1]) c1 = load8_f32(srow1 + mt * 16 + half8);
      v8f d1 = wmma_f16(a, xb[1], c1);
      if (inwin[1]) store8_f32(srow1 + mt * 16 + half8, d1);
    }
  }

  // res conv + residual add (fp32 residual stream), dual f32/f16 store
#pragma unroll
  for (int mt = 0; mt < 2; ++mt) {
    v16h a = load_afrag(blob, OFF_RES + stage * 2 + mt);
#pragma unroll
    for (int j = 0; j < 2; ++j) {
      bool valid = p[j] < Lout;
      v8f z = {};
      v8f d = wmma_f16(a, xb[j], z);
      v8f r = load8_f32(xin32 + ((size_t)(n * POSCAP + p[j] + 1)) * 32 +
                        mt * 16 + half8);
      v8f o;
#pragma unroll
      for (int i = 0; i < 8; ++i) o[i] = valid ? (d[i] + r[i]) : 0.0f;
      store8_f32(o32[j] + mt * 16 + half8, o);
      store8_f16(o16[j] + mt * 16 + half8, o);
    }
  }
}

// ---------------- fused end: out = end2 @ relu(end1 @ relu(skip) + b1) + b2 --

__global__ void __launch_bounds__(256)
end_kernel(const float* __restrict__ skipwin, const _Float16* __restrict__ blob,
           const float* __restrict__ b1v, const float* __restrict__ b2v,
           float* __restrict__ out) {
  int lane = threadIdx.x & 31;
  int flat = blockIdx.x * (blockDim.x >> 5) + (threadIdx.x >> 5);
  int tile = flat & 255;
  int n = flat >> 8;
  if (n >= NB) return;
  int col = lane & 15, half8 = ((lane >> 4) & 1) * 8;
  int w = tile * 16 + col;

  const float* srow = skipwin + ((size_t)n * OUTLEN + w) * SC;
  v16h bf[8];
#pragma unroll
  for (int kg = 0; kg < 8; ++kg) bf[kg] = bfrag_from_f32_relu(srow + kg * 32);

  // end1 + relu, converting D tiles to B fragments pairwise on the fly
  v16h b2[8];
#pragma unroll
  for (int kg2 = 0; kg2 < 8; ++kg2) {
    v8f dpair[2];
#pragma unroll
    for (int j = 0; j < 2; ++j) {
      int mt = kg2 * 2 + j;
      v8f c = load8_f32(b1v + mt * 16 + half8);  // bias in D layout
#pragma unroll
      for (int kg = 0; kg < 8; ++kg)
        c = wmma_f16(load_afrag(blob, OFF_END1 + mt * 8 + kg), bf[kg], c);
#pragma unroll
      for (int i = 0; i < 8; ++i) c[i] = fmaxf(c[i], 0.0f);
      dpair[j] = c;
    }
    b2[kg2] = dpair_to_bfrag(dpair[0], dpair[1]);
  }

  // end2 + bias, store (n*4096 + w, class)
#pragma unroll 4
  for (int mt = 0; mt < 16; ++mt) {
    v8f c = load8_f32(b2v + mt * 16 + half8);
#pragma unroll
    for (int kg = 0; kg < 8; ++kg)
      c = wmma_f16(load_afrag(blob, OFF_END2 + mt * 8 + kg), b2[kg], c);
    store8_f32(out + ((size_t)n * OUTLEN + w) * NCLASS + mt * 16 + half8, c);
  }
}

// ---------------- host ----------------

extern "C" void kernel_launch(void* const* d_in, const int* in_sizes, int n_in,
                              void* d_out, int out_size, void* d_ws,
                              size_t ws_size, hipStream_t stream) {
  (void)in_sizes; (void)n_in; (void)out_size; (void)ws_size;
  const float* input    = (const float*)d_in[0];
  const float* start_w  = (const float*)d_in[1];
  const float* filter_w = (const float*)d_in[2];
  const float* gate_w   = (const float*)d_in[3];
  const float* res_w    = (const float*)d_in[4];
  const float* skip_w   = (const float*)d_in[5];
  const float* end1_w   = (const float*)d_in[6];
  const float* end1_b   = (const float*)d_in[7];
  const float* end2_w   = (const float*)d_in[8];
  const float* end2_b   = (const float*)d_in[9];
  float* out = (float*)d_out;

  char* ws = (char*)d_ws;
  size_t off = 0;
  auto alloc = [&](size_t bytes) -> void* {
    off = (off + 255) & ~(size_t)255;
    void* p = ws + off;
    off += bytes;
    return p;
  };
  _Float16* in16  = (_Float16*)alloc((size_t)NB * LIN * NCLASS * 2);
  float*    xA32  = (float*)   alloc((size_t)NB * POSCAP * 32 * 4);
  float*    xB32  = (float*)   alloc((size_t)NB * POSCAP * 32 * 4);
  _Float16* xA16  = (_Float16*)alloc((size_t)NB * POSCAP * 32 * 2);
  _Float16* xB16  = (_Float16*)alloc((size_t)NB * POSCAP * 32 * 2);
  float*    skipw = (float*)   alloc((size_t)NB * OUTLEN * SC * 4);
  _Float16* blob  = (_Float16*)alloc((size_t)NTILES * 512 * 2);

  hipMemsetAsync(skipw, 0, (size_t)NB * OUTLEN * SC * 4, stream);
  prep_weights<<<NTILES, 32, 0, stream>>>(start_w, filter_w, gate_w, res_w,
                                          skip_w, end1_w, end2_w, blob);
  int total = NB * NCLASS * LIN;
  prep_input<<<(total + 255) / 256, 256, 0, stream>>>(input, in16, total);

  start_kernel<<<514, 256, 0, stream>>>(in16, blob, xA32, xA16);

  int L = LIN;
  float *c32 = xA32, *n32 = xB32;
  _Float16 *c16 = xA16, *n16 = xB16;
  for (int i = 0; i < NSTAGES; ++i) {
    int d = 1 << (i % 10);
    int pad = (d - (L % d)) % d;          // dilate(): pad to multiple of d
    int Lout = L + pad - 1;               // k=2 conv shrinks by 1
    stage_kernel<<<257, 256, 0, stream>>>(c32, c16, n32, n16, skipw, blob, i,
                                          Lout);
    float* t32 = c32; c32 = n32; n32 = t32;
    _Float16* t16 = c16; c16 = n16; n16 = t16;
    L = Lout;
  }

  end_kernel<<<256, 256, 0, stream>>>(skipw, blob, end1_b, end2_b, out);
}
